// ScaledDotProductAttention_60361470378664
// MI455X (gfx1250) — matrix operand order repaired and checked by reading, not compiled, not run on hardware
//
#include <hip/hip_runtime.h>

// ---------------------------------------------------------------------------
// Fused scaled-dot-product attention (materializing attn) for MI455X/gfx1250
//   out  = softmax(mask ? -1e9 : QK^T/sqrt(D)) @ V        [B,S,D] f32
//   attn = the softmax probabilities                      [B,S,S] f32
// B=32, S=2048, D=128.  Math in bf16 WMMA (v_wmma_f32_16x16x32_bf16), f32 accum.
//
// v2: P·V phase now stages V chunks through LDS with an explicit transpose
//     (coalesced float4 global loads + packed bf16 converts + ds_store_b128),
//     so the WMMA B operand is built from two contiguous ds_load_b128 per tile
//     instead of 16 strided scalar global gathers.
// ---------------------------------------------------------------------------

typedef __attribute__((ext_vector_type(16))) __bf16 v16bf;
typedef __attribute__((ext_vector_type(8)))  float  v8f;

#define B_  32
#define S_  2048
#define D_  128
#define QT  16                     // q rows per workgroup
#define NWAVE 8
#define KSLAB 256                  // k columns handled per wave
#define VTP 40                     // padded row stride (bf16) for transposed V
#define MASK_BIAS  (-1e9f)
#define INV_SCALE  0.08838834764831845f   // 1/sqrt(128)

union AVec { v16bf v; uint4 u[2]; };
union H8   { __bf16 h[8]; uint4 u; };

__global__ __launch_bounds__(256, 1)
void attn_fused_kernel(const float* __restrict__ Q,
                       const float* __restrict__ K,
                       const float* __restrict__ V,
                       const unsigned char* __restrict__ Msk,
                       float* __restrict__ Out,     // [B,S,D]
                       float* __restrict__ Attn)    // [B,S,S]
{
    // LDS: 64KB prob stage + 80KB transposed-V stage + 8KB out accum + scratch
    __shared__ __bf16 pb[QT][S_];            // probabilities (bf16) for P·V
    __shared__ __bf16 vt[NWAVE][D_][VTP];    // per-wave transposed V chunk
    __shared__ float  outacc[QT][D_];        // cross-wave output accumulator
    __shared__ float  wsred[NWAVE][QT];      // per-wave row stats (max, then sum)

    const int tid  = threadIdx.x;
    const int wave = tid >> 5;
    const int lane = tid & 31;
    const int l16  = lane & 15;
    const int half = lane >> 4;

    const int b  = blockIdx.x / (S_ / QT);
    const int qt = blockIdx.x % (S_ / QT);
    const int q0 = qt * QT;
    const int kw0 = wave * KSLAB;            // this wave's key-column slab

    // zero the output accumulator (consumed after later barriers)
    #pragma unroll
    for (int i = 0; i < (QT * D_) / 256; ++i)
        (&outacc[0][0])[i * 256 + tid] = 0.0f;

    // ------------------------------------------------------------------
    // Phase A: load Q tile (16 x 128) as 4 bf16 A-layout chunks (K=32 each)
    //   A-layout (16x32 bf16): lanes0-15 row M=lane, elems0..15 -> K=0..15,
    //   lanes16-31 hold K=16..31 (contiguous, matching B).
    // ------------------------------------------------------------------
    const float* qrow = Q + ((size_t)b * S_ + q0 + l16) * D_;
    v16bf aq[4];
    #pragma unroll
    for (int dc = 0; dc < 4; ++dc) {
        const int dbase = dc * 32;
        const int o0 = dbase + (half ? 16 : 0);
        const int o1 = dbase + 8 + (half ? 16 : 0);
        #pragma unroll
        for (int j = 0; j < 8; ++j) {
            aq[dc][j]     = (__bf16)qrow[o0 + j];
            aq[dc][j + 8] = (__bf16)qrow[o1 + j];
        }
    }

    // ------------------------------------------------------------------
    // Phase B: scores for this wave's 16 key tiles (16 keys each).
    //   B-layout (32x16 bf16): lane n (0..15) = column n (key row k0+n),
    //   lanes0-15 hold Kdim 0..15, lanes16-31 hold Kdim 16..31 (contig in D).
    // ------------------------------------------------------------------
    v8f sreg[16];
    #pragma unroll
    for (int kt = 0; kt < 16; ++kt) {
        const int k0 = kw0 + kt * 16;
        const float* krow = K + ((size_t)b * S_ + k0 + l16) * D_;
        v8f acc = {};
        #pragma unroll
        for (int dc = 0; dc < 4; ++dc) {
            const int dbase = dc * 32 + half * 16;
            v16bf bk;
            #pragma unroll
            for (int j = 0; j < 16; ++j)
                bk[j] = (__bf16)krow[dbase + j];
            acc = __builtin_amdgcn_wmma_f32_16x16x32_bf16(
                false, aq[dc], false, bk, (short)0, acc, false, false);
        }
        // scale + mask  (C-layout: VGPR r, lane -> row r+8*half, col l16)
        #pragma unroll
        for (int r = 0; r < 8; ++r) {
            const int gq = q0 + r + 8 * half;
            const int gk = k0 + l16;
            const unsigned char m = Msk[((size_t)b * S_ + gq) * S_ + gk];
            const float s = acc[r] * INV_SCALE;
            acc[r] = m ? MASK_BIAS : s;
        }
        sreg[kt] = acc;
    }

    // ------------------------------------------------------------------
    // Phase C: row max (per wave -> cross-wave via LDS)
    // ------------------------------------------------------------------
    float pm[8];
    #pragma unroll
    for (int r = 0; r < 8; ++r) {
        pm[r] = -3.4e38f;
        #pragma unroll
        for (int kt = 0; kt < 16; ++kt)
            pm[r] = fmaxf(pm[r], sreg[kt][r]);
        #pragma unroll
        for (int i = 1; i < 16; i <<= 1)
            pm[r] = fmaxf(pm[r], __shfl_xor(pm[r], i, 16));
    }
    if (l16 == 0) {
        #pragma unroll
        for (int r = 0; r < 8; ++r)
            wsred[wave][8 * half + r] = pm[r];
    }
    __syncthreads();
    float gm[8];
    #pragma unroll
    for (int r = 0; r < 8; ++r) {
        float m = wsred[0][8 * half + r];
        #pragma unroll
        for (int w = 1; w < NWAVE; ++w)
            m = fmaxf(m, wsred[w][8 * half + r]);
        gm[r] = m;
    }
    __syncthreads();

    // ------------------------------------------------------------------
    // Phase D: exp + row sum (per wave -> cross-wave via LDS)
    // ------------------------------------------------------------------
    float ps[8];
    #pragma unroll
    for (int r = 0; r < 8; ++r) ps[r] = 0.0f;
    #pragma unroll
    for (int kt = 0; kt < 16; ++kt) {
        #pragma unroll
        for (int r = 0; r < 8; ++r) {
            const float e = __expf(sreg[kt][r] - gm[r]);
            sreg[kt][r] = e;
            ps[r] += e;
        }
    }
    #pragma unroll
    for (int r = 0; r < 8; ++r) {
        #pragma unroll
        for (int i = 1; i < 16; i <<= 1)
            ps[r] += __shfl_xor(ps[r], i, 16);
    }
    if (l16 == 0) {
        #pragma unroll
        for (int r = 0; r < 8; ++r)
            wsred[wave][8 * half + r] = ps[r];
    }
    __syncthreads();
    float inv[8];
    #pragma unroll
    for (int r = 0; r < 8; ++r) {
        float s = 0.0f;
        #pragma unroll
        for (int w = 0; w < NWAVE; ++w)
            s += wsred[w][8 * half + r];
        inv[r] = 1.0f / s;
    }

    // ------------------------------------------------------------------
    // Phase E: probabilities -> global attn (f32) + LDS stage (bf16)
    // ------------------------------------------------------------------
    #pragma unroll
    for (int kt = 0; kt < 16; ++kt) {
        const int k0 = kw0 + kt * 16;
        #pragma unroll
        for (int r = 0; r < 8; ++r) {
            const int row = r + 8 * half;
            const float p = sreg[kt][r] * inv[r];
            Attn[((size_t)b * S_ + q0 + row) * S_ + k0 + l16] = p;
            pb[row][k0 + l16] = (__bf16)p;
        }
    }
    __syncthreads();

    // ------------------------------------------------------------------
    // Phase F: O_partial = P_slab (16 x 256) @ V_slab (256 x 128)
    //   Per 32-key chunk: stage V(32 x 128) transposed into LDS vt[d][k]
    //   (coalesced float4 loads, packed bf16 cvts, ds_store_b128), then
    //   A from pb and B from vt are both contiguous 16B LDS loads.
    // ------------------------------------------------------------------
    v8f oacc[8];
    #pragma unroll
    for (int t = 0; t < 8; ++t) oacc[t] = (v8f){};

    #pragma unroll 1
    for (int c = 0; c < 8; ++c) {                 // 8 K-chunks of 32
        const int kA0 = kw0 + c * 32;

        // ---- stage: vt[wave][d][kk] = bf16(V[kA0+kk][d]) ----
        #pragma unroll
        for (int ko = 0; ko < 4; ++ko) {          // k-octet per iteration
            const int dq = lane;                  // d-quad = 4*lane
            float4 rowv[8];
            #pragma unroll
            for (int j = 0; j < 8; ++j)
                rowv[j] = *(const float4*)&V[((size_t)b * S_ + kA0 + ko * 8 + j) * D_ + dq * 4];
            #pragma unroll
            for (int e = 0; e < 4; ++e) {         // d = 4*lane + e
                H8 t8;
                #pragma unroll
                for (int j = 0; j < 8; ++j)
                    t8.h[j] = (__bf16)(((const float*)&rowv[j])[e]);
                *(uint4*)&vt[wave][4 * dq + e][ko * 8] = t8.u;
            }
        }

        // ---- A tile: P rows from pb (exact A-layout, two 16B ds loads) ----
        AVec ap;
        ap.u[0] = *(const uint4*)&pb[l16][kA0 + half * 8];
        ap.u[1] = *(const uint4*)&pb[l16][kA0 + 16 + half * 8];

        // ---- B tiles: contiguous rows of transposed V ----
        #pragma unroll
        for (int t = 0; t < 8; ++t) {             // 8 output d-tiles of 16
            AVec bv;
            const __bf16* vrow = &vt[wave][t * 16 + l16][half * 8];
            bv.u[0] = *(const uint4*)(vrow);
            bv.u[1] = *(const uint4*)(vrow + 16);
            oacc[t] = __builtin_amdgcn_wmma_f32_16x16x32_bf16(
                false, ap.v, false, bv.v, (short)0, oacc[t], false, false);
        }
    }

    // Phase G: merge partials across waves (ds_add_f32 atomics)
    #pragma unroll
    for (int t = 0; t < 8; ++t) {
        #pragma unroll
        for (int r = 0; r < 8; ++r)
            atomicAdd(&outacc[r + 8 * half][t * 16 + l16], oacc[t][r]);
    }
    __syncthreads();

    // Phase H: coalesced store of the 16x128 output tile
    #pragma unroll
    for (int i = 0; i < (QT * D_) / 256; ++i) {
        const int idx = i * 256 + tid;
        const int row = idx / D_;
        const int col = idx % D_;
        Out[((size_t)b * S_ + q0 + row) * D_ + col] = outacc[row][col];
    }
}

extern "C" void kernel_launch(void* const* d_in, const int* in_sizes, int n_in,
                              void* d_out, int out_size, void* d_ws, size_t ws_size,
                              hipStream_t stream) {
    const float* q = (const float*)d_in[0];
    const float* k = (const float*)d_in[1];
    const float* v = (const float*)d_in[2];
    const unsigned char* mask = (const unsigned char*)d_in[3];  // jax bool = 1 byte

    float* out  = (float*)d_out;                           // [B,S,D] first
    float* attn = out + (size_t)B_ * S_ * D_;              // then [B,S,S]

    dim3 grid(B_ * (S_ / QT));   // 32 * 128 = 4096 workgroups
    dim3 block(256);             // 8 waves (wave32)
    attn_fused_kernel<<<grid, block, 0, stream>>>(q, k, v, mask, out, attn);
}
